// CausalSelfAttention_80814104641605
// MI455X (gfx1250) — compile-verified, matrix operand order checked by reading, not run on hardware
//
#include <hip/hip_runtime.h>
#include <hip/hip_bf16.h>

typedef __attribute__((ext_vector_type(16))) __bf16 v16bf;
typedef __attribute__((ext_vector_type(8)))  __bf16 v8bf;
typedef __attribute__((ext_vector_type(8)))  float  v8f;

#define BI 2
#define TI 2048
#define CI 1024
#define HI 16
#define HDI 64
#define MTOT (BI*TI)      /* 4096 */
#define NQKV (3*CI)       /* 3072 */

// ---- helpers -------------------------------------------------------------

static __device__ __forceinline__ v16bf combine8(v8bf lo, v8bf hi) {
  return __builtin_shufflevector(lo, hi, 0,1,2,3,4,5,6,7,8,9,10,11,12,13,14,15);
}

// Loads the per-lane A/B operand half-chunks: p[0..7] and p[16..23]
static __device__ __forceinline__ v16bf load_op16(const __bf16* p) {
  v8bf lo = *reinterpret_cast<const v8bf*>(p);
  v8bf hi = *reinterpret_cast<const v8bf*>(p + 16);
  return combine8(lo, hi);
}

static __device__ __forceinline__ v8f wmma_bf16(v16bf a, v16bf b, v8f c) {
  return __builtin_amdgcn_wmma_f32_16x16x32_bf16(false, a, false, b, (short)0, c,
                                                 false, false);
}

// ---- precision / layout prep kernels ------------------------------------

__global__ void f32_to_bf16_kernel(const float* __restrict__ in,
                                   __bf16* __restrict__ out, int n) {
  int i = (blockIdx.x * blockDim.x + threadIdx.x) * 4;
  if (i + 3 < n) {
    float4 v = *reinterpret_cast<const float4*>(in + i);
    out[i + 0] = (__bf16)v.x;
    out[i + 1] = (__bf16)v.y;
    out[i + 2] = (__bf16)v.z;
    out[i + 3] = (__bf16)v.w;
  }
}

// out[c][r] = (bf16) in[r][c]   (in: rows_in x cols_in row-major)
__global__ void transpose_to_bf16_kernel(const float* __restrict__ in,
                                         __bf16* __restrict__ out,
                                         int rows_in, int cols_in) {
  int c = blockIdx.x * 16 + threadIdx.x;
  int r = blockIdx.y * 16 + threadIdx.y;
  if (c < cols_in && r < rows_in)
    out[(size_t)c * rows_in + r] = (__bf16)in[(size_t)r * cols_in + c];
}

// ---- QKV projection GEMM: [4096x1024] @ [1024x3072] ----------------------
// Bt is w_attn transposed: [3072][1024]. Each wave -> 16x64 tile of output,
// scattered to q [B,H,T,64], k [B,H,T,64], v transposed [B,H,64,T].

__global__ __launch_bounds__(256) void qkv_gemm_kernel(
    const __bf16* __restrict__ A,   // [4096][1024]
    const __bf16* __restrict__ Bt,  // [3072][1024]
    __bf16* __restrict__ qb, __bf16* __restrict__ kb, __bf16* __restrict__ vt) {
  int wave = (blockIdx.x * blockDim.x + threadIdx.x) >> 5;
  int lane = threadIdx.x & 31;
  int half = lane >> 4, lm = lane & 15;
  int cb = half * 8;

  int mtile = wave & (MTOT / 16 - 1);   // 256 m-tiles
  int n64   = wave >> 8;                // 48 n-supertiles
  int m0 = mtile * 16, n0 = n64 * 64;

  const __bf16* arow = A + (size_t)(m0 + lm) * CI;
  const __bf16* brow[4];
#pragma unroll
  for (int nt = 0; nt < 4; ++nt)
    brow[nt] = Bt + (size_t)(n0 + nt * 16 + lm) * CI;

  v8f acc[4] = {};
  for (int k0 = 0; k0 < CI; k0 += 32) {
    v16bf a = load_op16(arow + k0 + cb);
#pragma unroll
    for (int nt = 0; nt < 4; ++nt) {
      v16bf b = load_op16(brow[nt] + k0 + cb);
      acc[nt] = wmma_bf16(a, b, acc[nt]);
    }
  }

#pragma unroll
  for (int nt = 0; nt < 4; ++nt) {
    int col = n0 + nt * 16 + lm;           // [0,3072)
    int which = col >> 10;                 // 0=Q 1=K 2=V
    int cj = col & (CI - 1);
    int h = cj >> 6, d = cj & 63;
#pragma unroll
    for (int r = 0; r < 8; ++r) {
      int row = m0 + r + 8 * half;         // token [0,4096)
      int bb = row >> 11, t = row & (TI - 1);
      size_t bh = (size_t)(bb * HI + h);
      __bf16 v = (__bf16)acc[nt][r];
      if (which == 0)      qb[(bh * TI + t) * HDI + d] = v;
      else if (which == 1) kb[(bh * TI + t) * HDI + d] = v;
      else                 vt[(bh * HDI + d) * TI + t] = v;
    }
  }
}

// ---- flash attention: one wave per (b,h,16-row q tile) -------------------

__global__ __launch_bounds__(32) void attn_kernel(
    const __bf16* __restrict__ qb, const __bf16* __restrict__ kbuf,
    const __bf16* __restrict__ vt, __bf16* __restrict__ yb) {
  __shared__ __align__(16) __bf16 pls[16 * 32];   // P tile staging (per wave)

  int bid = blockIdx.x;                 // B*H*(T/16) = 4096
  int qtile = bid & (TI / 16 - 1);      // 128
  int bh = bid >> 7;                    // 32
  int lane = threadIdx.x & 31;
  int half = lane >> 4, lm = lane & 15;
  int cb = half * 8;
  int qbase = qtile * 16;

  const __bf16* Q  = qb   + (size_t)bh * TI * HDI;
  const __bf16* Km = kbuf + (size_t)bh * TI * HDI;
  const __bf16* Vt = vt   + (size_t)bh * HDI * TI;

  const __bf16* qr = Q + (size_t)(qbase + lm) * HDI;
  v16bf aq0 = load_op16(qr + cb);        // d = 0..31 chunk
  v16bf aq1 = load_op16(qr + 32 + cb);   // d = 32..63 chunk

  float m[8], l[8];
  v8f accO[4] = {};
#pragma unroll
  for (int r = 0; r < 8; ++r) { m[r] = -1e30f; l[r] = 0.f; }

  const float scale = 0.125f;            // 1/sqrt(64)
  int kend = qbase + 16;                 // causal: keys <= q row
  for (int k0 = 0; k0 < kend; k0 += 32) {
    // ---- S = Q K^T for 32 keys (two 16-wide n-tiles) ----
    v8f S[2];
#pragma unroll
    for (int s = 0; s < 2; ++s) {
      int key = k0 + s * 16 + lm;        // always < T
      const __bf16* kr = Km + (size_t)key * HDI;
      v16bf bk0 = load_op16(kr + cb);
      v16bf bk1 = load_op16(kr + 32 + cb);
      v8f z = {};
      z = wmma_bf16(aq0, bk0, z);
      S[s] = wmma_bf16(aq1, bk1, z);
    }
    // ---- scale + causal mask, per-row online softmax ----
    float p[2][8], sm[8];
#pragma unroll
    for (int r = 0; r < 8; ++r) {
      int qrow = qbase + r + 8 * half;
      float s0 = (k0 + lm      <= qrow) ? S[0][r] * scale : -1e30f;
      float s1 = (k0 + 16 + lm <= qrow) ? S[1][r] * scale : -1e30f;
      p[0][r] = s0; p[1][r] = s1;
      sm[r] = fmaxf(s0, s1);
    }
#pragma unroll
    for (int msk = 1; msk < 16; msk <<= 1)
#pragma unroll
      for (int r = 0; r < 8; ++r)
        sm[r] = fmaxf(sm[r], __shfl_xor(sm[r], msk, 32));

    float rs[8];
#pragma unroll
    for (int r = 0; r < 8; ++r) {
      float mn = fmaxf(m[r], sm[r]);
      float alpha = __expf(m[r] - mn);
      m[r] = mn;
      float p0 = __expf(p[0][r] - mn);
      float p1 = __expf(p[1][r] - mn);
      p[0][r] = p0; p[1][r] = p1;
      rs[r] = p0 + p1;
      l[r] *= alpha;
#pragma unroll
      for (int nt = 0; nt < 4; ++nt) accO[nt][r] *= alpha;
    }
#pragma unroll
    for (int msk = 1; msk < 16; msk <<= 1)
#pragma unroll
      for (int r = 0; r < 8; ++r) rs[r] += __shfl_xor(rs[r], msk, 32);
#pragma unroll
    for (int r = 0; r < 8; ++r) l[r] += rs[r];

    // ---- P: C/D layout -> A-operand layout via LDS ----
#pragma unroll
    for (int s = 0; s < 2; ++s)
#pragma unroll
      for (int r = 0; r < 8; ++r)
        pls[(r + 8 * half) * 32 + s * 16 + lm] = (__bf16)p[s][r];
    asm volatile("s_wait_dscnt 0" ::: "memory");
    v16bf ap = load_op16(&pls[lm * 32 + cb]);

    // ---- accO += P @ V (V pre-transposed: rows are dims, contiguous keys)
#pragma unroll
    for (int nt = 0; nt < 4; ++nt) {
      const __bf16* vr = Vt + (size_t)(nt * 16 + lm) * TI + k0;
      v16bf bv = load_op16(vr + cb);
      accO[nt] = wmma_bf16(ap, bv, accO[nt]);
    }
    asm volatile("s_wait_dscnt 0" ::: "memory"); // LDS reads done before rewrite
  }

  // ---- write y (bf16, [4096][1024]) ----
  int h = bh & (HI - 1), bb = bh >> 4;
#pragma unroll
  for (int r = 0; r < 8; ++r) {
    int t = qbase + r + 8 * half;
    size_t row = (size_t)bb * TI + t;
    float inv = 1.0f / l[r];
#pragma unroll
    for (int nt = 0; nt < 4; ++nt)
      yb[row * CI + h * HDI + nt * 16 + lm] = (__bf16)(accO[nt][r] * inv);
  }
}

// ---- output projection GEMM: [4096x1024] @ [1024x1024] -> f32 out -------

__global__ __launch_bounds__(256) void proj_gemm_kernel(
    const __bf16* __restrict__ A,   // [4096][1024]
    const __bf16* __restrict__ Bt,  // [1024][1024] (w_proj^T)
    float* __restrict__ out) {
  int wave = (blockIdx.x * blockDim.x + threadIdx.x) >> 5;
  int lane = threadIdx.x & 31;
  int half = lane >> 4, lm = lane & 15;
  int cb = half * 8;

  int mtile = wave & (MTOT / 16 - 1);
  int n64   = wave >> 8;                // 16 n-supertiles
  int m0 = mtile * 16, n0 = n64 * 64;

  const __bf16* arow = A + (size_t)(m0 + lm) * CI;
  const __bf16* brow[4];
#pragma unroll
  for (int nt = 0; nt < 4; ++nt)
    brow[nt] = Bt + (size_t)(n0 + nt * 16 + lm) * CI;

  v8f acc[4] = {};
  for (int k0 = 0; k0 < CI; k0 += 32) {
    v16bf a = load_op16(arow + k0 + cb);
#pragma unroll
    for (int nt = 0; nt < 4; ++nt) {
      v16bf b = load_op16(brow[nt] + k0 + cb);
      acc[nt] = wmma_bf16(a, b, acc[nt]);
    }
  }
#pragma unroll
  for (int nt = 0; nt < 4; ++nt) {
    int col = n0 + nt * 16 + lm;
#pragma unroll
    for (int r = 0; r < 8; ++r) {
      int row = m0 + r + 8 * half;
      out[(size_t)row * CI + col] = acc[nt][r];
    }
  }
}

// ---- launch --------------------------------------------------------------

extern "C" void kernel_launch(void* const* d_in, const int* in_sizes, int n_in,
                              void* d_out, int out_size, void* d_ws, size_t ws_size,
                              hipStream_t stream) {
  const float* x      = (const float*)d_in[0];   // [2,2048,1024]
  const float* w_attn = (const float*)d_in[1];   // [1024,3072]
  const float* w_proj = (const float*)d_in[2];   // [1024,1024]
  float* out = (float*)d_out;                    // [2,2048,1024] f32

  char* ws = (char*)d_ws;
  size_t o = 0;
  __bf16* xb  = (__bf16*)(ws + o); o += (size_t)MTOT * CI * 2;   //  8 MB
  __bf16* wat = (__bf16*)(ws + o); o += (size_t)NQKV * CI * 2;   //  6 MB
  __bf16* wpt = (__bf16*)(ws + o); o += (size_t)CI * CI * 2;     //  2 MB
  __bf16* qb  = (__bf16*)(ws + o); o += (size_t)MTOT * CI * 2;   //  8 MB
  __bf16* kb  = (__bf16*)(ws + o); o += (size_t)MTOT * CI * 2;   //  8 MB
  __bf16* vt  = (__bf16*)(ws + o); o += (size_t)MTOT * CI * 2;   //  8 MB
  __bf16* yb  = (__bf16*)(ws + o); o += (size_t)MTOT * CI * 2;   //  8 MB
  (void)ws_size; (void)in_sizes; (void)n_in; (void)out_size;

  f32_to_bf16_kernel<<<(MTOT * CI / 4 + 255) / 256, 256, 0, stream>>>(x, xb, MTOT * CI);
  transpose_to_bf16_kernel<<<dim3(NQKV / 16, CI / 16), dim3(16, 16), 0, stream>>>(
      w_attn, wat, CI, NQKV);
  transpose_to_bf16_kernel<<<dim3(CI / 16, CI / 16), dim3(16, 16), 0, stream>>>(
      w_proj, wpt, CI, CI);

  // 256 m-tiles * 48 n64-tiles = 12288 waves / 8 per block
  qkv_gemm_kernel<<<(MTOT / 16) * (NQKV / 64) / 8, 256, 0, stream>>>(xb, wat, qb, kb, vt);

  attn_kernel<<<BI * HI * (TI / 16), 32, 0, stream>>>(qb, kb, vt, yb);

  // 256 m-tiles * 16 n64-tiles = 4096 waves / 8 per block
  proj_gemm_kernel<<<(MTOT / 16) * (CI / 64) / 8, 256, 0, stream>>>(yb, wpt, out);
}
